// TromptConv_28604482191965
// MI455X (gfx1250) — matrix operand order repaired and checked by reading, not compiled, not run on hardware
//
#include <hip/hip_runtime.h>
#include <hip/hip_bf16.h>

// ---------- CDNA5 WMMA types ----------
typedef __attribute__((ext_vector_type(16))) __bf16 v16bf;
typedef __attribute__((ext_vector_type(8)))  float  v8f;
typedef __attribute__((ext_vector_type(4)))  unsigned int v4u;
typedef __attribute__((ext_vector_type(8)))  int v8i_;
typedef __attribute__((ext_vector_type(4)))  int v4i_;

union Frag {
    v16bf bf;
    uint4 q[2];
    unsigned short us[16];
};

static __device__ __forceinline__ unsigned short f2bf(float f) {
    union { float f; unsigned u; } v; v.f = f;
    unsigned u = v.u;
    unsigned r = u + 0x7FFFu + ((u >> 16) & 1u);   // round-to-nearest-even
    return (unsigned short)(r >> 16);
}

static __device__ __forceinline__ v8f wmma_bf16(const Frag& a, const Frag& b, v8f c) {
    return __builtin_amdgcn_wmma_f32_16x16x32_bf16(
        /*neg_a=*/false, a.bf, /*neg_b=*/false, b.bf,
        /*c_mod=*/(short)0, c, /*reuse_a=*/false, /*reuse_b=*/false);
}

// Problem constants
#define B_  512
#define C_  64
#define H_  128
#define P_  16
#define EPS 1e-5f

// ---------- Kernel 0: layernorm emb_column / emb_prompt -> bf16 (+ep f32) ----------
__global__ __launch_bounds__(128)
void k_prep_ln(const float* __restrict__ emb_col, const float* __restrict__ emb_pr,
               const float* __restrict__ lnc_w, const float* __restrict__ lnc_b,
               const float* __restrict__ lnp_w, const float* __restrict__ lnp_b,
               unsigned short* __restrict__ ec_bf, unsigned short* __restrict__ ep_bf,
               float* __restrict__ ep_f) {
    __shared__ float red[128];
    int row = blockIdx.x;        // 0..63 -> emb_column, 64..79 -> emb_prompt
    int t = threadIdx.x;
    bool isCol = row < C_;
    float v = isCol ? emb_col[row * H_ + t] : emb_pr[(row - C_) * H_ + t];

    red[t] = v; __syncthreads();
    #pragma unroll
    for (int s = 64; s > 0; s >>= 1) { if (t < s) red[t] += red[t + s]; __syncthreads(); }
    float mu = red[0] * (1.0f / H_);
    __syncthreads();
    float d = v - mu;
    red[t] = d * d; __syncthreads();
    #pragma unroll
    for (int s = 64; s > 0; s >>= 1) { if (t < s) red[t] += red[t + s]; __syncthreads(); }
    float var = red[0] * (1.0f / H_);
    float rs = rsqrtf(var + EPS);

    if (isCol) {
        float o = d * rs * lnc_w[t] + lnc_b[t];
        ec_bf[row * H_ + t] = f2bf(o);
    } else {
        float o = d * rs * lnp_w[t] + lnp_b[t];
        ep_bf[(row - C_) * H_ + t] = f2bf(o);
        ep_f[(row - C_) * H_ + t] = o;
    }
}

// ---------- Kernel 0b: lin_w f32 -> bf16 ----------
__global__ __launch_bounds__(256)
void k_conv_lw(const float* __restrict__ lw, unsigned short* __restrict__ lw_bf) {
    int i = blockIdx.x * 256 + threadIdx.x;   // 128*256 = 32768 elements
    lw_bf[i] = f2bf(lw[i]);
}

// ---------- Main fused kernel: one block per batch ----------
__global__ __launch_bounds__(256)
void k_main(const float* __restrict__ x, const float* __restrict__ x_prompt,
            const float* __restrict__ lin_b,
            const float* __restrict__ expand_w, const float* __restrict__ gn_w,
            const float* __restrict__ gn_b,
            const unsigned short* __restrict__ ec_bf,
            const unsigned short* __restrict__ ep_bf,
            const float* __restrict__ ep_f,
            const unsigned short* __restrict__ lw_bf,
            float* __restrict__ out) {
    __shared__ float xl[C_ * H_];                 // 32 KB : x[b] f32 (TDM destination)
    __shared__ unsigned short catb[P_ * 264];     // cat = [ep | x_prompt[b]] bf16, padded stride
    __shared__ unsigned short hatb[P_ * 136];     // hat bf16, padded stride
    __shared__ float msm[P_ * C_];                // logits -> softmax probs
    __shared__ unsigned short mb[P_ * 72];        // softmax probs bf16, padded stride
    __shared__ float redsh[16];
    __shared__ float alpha_s[P_], beta_s[P_], ew_s[P_];

    const int tid = threadIdx.x;
    const int b = blockIdx.x;
    const int lane = tid & 31;
    const int wv = tid >> 5;          // 8 waves
    const int col = lane & 15;        // tile column / A-row id
    const int hi = lane >> 4;         // half-wave id

    const float* xg = x + (size_t)b * (C_ * H_);
    const float* xpg = x_prompt + (size_t)b * (P_ * H_);

    // Prefetch next batch's x (global_prefetch_b8 path)
    if (b + 1 < B_) __builtin_prefetch(x + (size_t)(b + 1) * (C_ * H_) + tid * 32, 0, 1);

    // ---- Stage x[b] into LDS via the Tensor Data Mover (wave 0 issues one DMA) ----
#if __has_builtin(__builtin_amdgcn_tensor_load_to_lds) && __has_builtin(__builtin_amdgcn_s_wait_tensorcnt)
    if (wv == 0) {
        unsigned long long ga = (unsigned long long)(uintptr_t)xg;   // 57-bit global byte addr
        unsigned lds_off = (unsigned)(uintptr_t)(&xl[0]);            // LDS byte offset (low 32 bits)
        // D# group 0: count=1(valid), lds_addr, global_addr[56:0], type=2 ("image")
        v4u g0 = { 1u,
                   lds_off,
                   (unsigned)(ga & 0xFFFFFFFFull),
                   (unsigned)((ga >> 32) & 0x1FFFFFFull) | (2u << 30) };
        // D# group 1: mask=0, data_size=4B(code 2), tensor_dim0=8192, tensor_dim1=1,
        //             tile_dim0=8192, tile_dim1=1, tensor_dim0_stride=8192
        v8i_ g1 = { (int)(2u << 16),                                   // data_size=2 (4 bytes)
                    (int)((8192u & 0xFFFFu) << 16),                    // tensor_dim0 lo16 -> [31:16]
                    (int)((8192u >> 16) | (1u << 16)),                 // tensor_dim0 hi16 | tensor_dim1 lo16
                    (int)(8192u << 16),                                // tensor_dim1 hi16 | tile_dim0=8192
                    (int)1,                                            // tile_dim1=1, tile_dim2=0
                    (int)8192,                                         // tensor_dim0_stride lo32
                    0, 0 };
        v4i_ gz4 = { 0, 0, 0, 0 };                                     // groups 2/3 unused (<=2D)
        v8i_ gz8 = { 0, 0, 0, 0, 0, 0, 0, 0 };                         // extra group (clang-23 6-arg form)
        __builtin_amdgcn_tensor_load_to_lds(g0, g1, gz4, gz4, gz8, 0);
        __builtin_amdgcn_s_wait_tensorcnt(0);                          // s_wait_tensorcnt 0
    }
#else
    {
        const float4* xg4 = (const float4*)xg;
        float4* xl4 = (float4*)xl;
        #pragma unroll
        for (int i = tid; i < (C_ * H_) / 4; i += 256) xl4[i] = xg4[i];
    }
#endif

    // ---- Build cat bf16: left = ln(emb_prompt), right = x_prompt[b] ----
    for (int e = tid; e < P_ * 256; e += 256) {
        int p = e >> 8, k = e & 255;
        unsigned short v = (k < 128) ? ep_bf[p * H_ + k] : f2bf(xpg[p * H_ + (k - 128)]);
        catb[p * 264 + k] = v;
    }
    __syncthreads();   // publishes TDM-filled xl + catb to all waves

    // ---- Phase A: hat[16,128] = cat[16,256] x lin_w^T, K=256 ----
    {
        const int h0 = wv * 16;
        v8f acc = {};
        #pragma unroll
        for (int kc = 0; kc < 8; ++kc) {
            const int kb = kc * 32;
            Frag a, bf;
            a.q[0] = *(const uint4*)(catb + col * 264 + kb + hi * 16);
            a.q[1] = *(const uint4*)(catb + col * 264 + kb + hi * 16 + 8);
            const uint4* bp = (const uint4*)(lw_bf + (size_t)(h0 + col) * 256 + kb + hi * 16);
            bf.q[0] = bp[0]; bf.q[1] = bp[1];
            acc = wmma_bf16(a, bf, acc);
        }
        const int h = h0 + col;
        const float lb = lin_b[h];
        #pragma unroll
        for (int r = 0; r < 8; ++r) {
            int p = r + hi * 8;
            float v = acc[r] + lb + ep_f[p * H_ + h] + xpg[p * H_ + h];
            hatb[p * 136 + h] = f2bf(v);
        }
    }
    __syncthreads();

    // ---- Phase B (waves 0-3): logits[16,64] = hat x ec^T, K=128 ----
    // ---- Waves 4-7 concurrently: GroupNorm scalar stats over x[b] ----
    if (wv < 4) {
        const int c0 = wv * 16;
        v8f acc = {};
        #pragma unroll
        for (int kc = 0; kc < 4; ++kc) {
            const int kb = kc * 32;
            Frag a, bf;
            a.q[0] = *(const uint4*)(hatb + col * 136 + kb + hi * 16);
            a.q[1] = *(const uint4*)(hatb + col * 136 + kb + hi * 16 + 8);
            const uint4* bp = (const uint4*)(ec_bf + (size_t)(c0 + col) * 128 + kb + hi * 16);
            bf.q[0] = bp[0]; bf.q[1] = bp[1];
            acc = wmma_bf16(a, bf, acc);
        }
        #pragma unroll
        for (int r = 0; r < 8; ++r) msm[(r + hi * 8) * C_ + c0 + col] = acc[r];
    } else {
        int st = tid - 128;
        float sp = 0.f, sn = 0.f, sp2 = 0.f, sn2 = 0.f;
        for (int e = st; e < C_ * H_; e += 128) {
            float v = xl[e];
            float rp = fmaxf(v, 0.f), rn = fmaxf(-v, 0.f);
            sp += rp; sn += rn; sp2 += rp * rp; sn2 += rn * rn;
        }
        #pragma unroll
        for (int off = 16; off > 0; off >>= 1) {
            sp  += __shfl_xor(sp, off);
            sn  += __shfl_xor(sn, off);
            sp2 += __shfl_xor(sp2, off);
            sn2 += __shfl_xor(sn2, off);
        }
        if (lane == 0) {
            int wi = wv - 4;
            redsh[wi * 4 + 0] = sp;  redsh[wi * 4 + 1] = sn;
            redsh[wi * 4 + 2] = sp2; redsh[wi * 4 + 3] = sn2;
        }
    }
    __syncthreads();

    // ---- Softmax (threads 0..15, one row each) + GroupNorm constants (thread 16) ----
    if (tid < 16) {
        int p = tid;
        float mx = -1e30f;
        for (int c = 0; c < C_; ++c) mx = fmaxf(mx, msm[p * C_ + c]);
        float s = 0.f;
        for (int c = 0; c < C_; ++c) s += __expf(msm[p * C_ + c] - mx);
        float is = 1.f / s;
        for (int c = 0; c < C_; ++c) mb[p * 72 + c] = f2bf(__expf(msm[p * C_ + c] - mx) * is);
    } else if (tid == 16) {
        float Sp  = redsh[0] + redsh[4] + redsh[8]  + redsh[12];
        float Sn  = redsh[1] + redsh[5] + redsh[9]  + redsh[13];
        float Sp2 = redsh[2] + redsh[6] + redsh[10] + redsh[14];
        float Sn2 = redsh[3] + redsh[7] + redsh[11] + redsh[15];
        const float inv = 1.0f / (8.0f * C_ * H_);   // (P/G)*C*H = 65536
        #pragma unroll
        for (int g = 0; g < 2; ++g) {
            float s1 = 0.f, s2 = 0.f;
            #pragma unroll
            for (int j = 0; j < 8; ++j) {
                float w = expand_w[g * 8 + j];
                s1 += (w >= 0.f) ? w * Sp : -w * Sn;
                s2 += w * w * ((w >= 0.f) ? Sp2 : Sn2);
            }
            float mu = s1 * inv;
            float var = s2 * inv - mu * mu;
            float rs = rsqrtf(var + EPS);
            #pragma unroll
            for (int j = 0; j < 8; ++j) {
                int p = g * 8 + j;
                float gw = gn_w[p];
                alpha_s[p] = rs * gw;
                beta_s[p]  = gn_b[p] - mu * rs * gw;
                ew_s[p]    = expand_w[p];
            }
        }
    }
    __syncthreads();

    // ---- Phase D: three fused [16x64]x[64x128] GEMMs (x, relu(x), relu(-x)) ----
    {
        const int h0 = wv * 16;
        const int h = h0 + col;
        v8f ax = {}, ap = {}, an = {};
        #pragma unroll
        for (int kc = 0; kc < 2; ++kc) {
            const int kb = kc * 32;
            Frag a, bx, bp, bn;
            a.q[0] = *(const uint4*)(mb + col * 72 + kb + hi * 16);
            a.q[1] = *(const uint4*)(mb + col * 72 + kb + hi * 16 + 8);
            const int cbase = kb + hi * 16;
            #pragma unroll
            for (int i = 0; i < 16; ++i) {
                float v = xl[(cbase + i) * H_ + h];
                bx.us[i] = f2bf(v);
                bp.us[i] = f2bf(fmaxf(v, 0.f));
                bn.us[i] = f2bf(fmaxf(-v, 0.f));
            }
            ax = wmma_bf16(a, bx, ax);
            ap = wmma_bf16(a, bp, ap);
            an = wmma_bf16(a, bn, an);
        }
        float* ob = out + (size_t)b * (P_ * H_);
        #pragma unroll
        for (int r = 0; r < 8; ++r) {
            int p = r + hi * 8;
            float w = ew_s[p];
            float sel = (w >= 0.f) ? w * ap[r] : -w * an[r];
            ob[p * H_ + h] = alpha_s[p] * sel + beta_s[p] + ax[r];
        }
    }
}

extern "C" void kernel_launch(void* const* d_in, const int* in_sizes, int n_in,
                              void* d_out, int out_size, void* d_ws, size_t ws_size,
                              hipStream_t stream) {
    (void)in_sizes; (void)n_in; (void)out_size; (void)ws_size;
    const float* x          = (const float*)d_in[0];
    const float* x_prompt   = (const float*)d_in[1];
    const float* emb_column = (const float*)d_in[2];
    const float* emb_prompt = (const float*)d_in[3];
    const float* lin_w      = (const float*)d_in[4];
    const float* lin_b      = (const float*)d_in[5];
    const float* ln_col_w   = (const float*)d_in[6];
    const float* ln_col_b   = (const float*)d_in[7];
    const float* ln_pr_w    = (const float*)d_in[8];
    const float* ln_pr_b    = (const float*)d_in[9];
    const float* expand_w   = (const float*)d_in[10];
    const float* gn_w       = (const float*)d_in[11];
    const float* gn_b       = (const float*)d_in[12];

    char* wsb = (char*)d_ws;
    unsigned short* ec_bf = (unsigned short*)(wsb);           // 64*128*2  = 16384
    unsigned short* ep_bf = (unsigned short*)(wsb + 16384);   // 16*128*2  = 4096
    float*          ep_f  = (float*)(wsb + 20480);            // 16*128*4  = 8192
    unsigned short* lw_bf = (unsigned short*)(wsb + 28672);   // 128*256*2 = 65536

    k_prep_ln<<<C_ + P_, 128, 0, stream>>>(emb_column, emb_prompt,
                                           ln_col_w, ln_col_b, ln_pr_w, ln_pr_b,
                                           ec_bf, ep_bf, ep_f);
    k_conv_lw<<<128, 256, 0, stream>>>(lin_w, lw_bf);
    k_main<<<B_, 256, 0, stream>>>(x, x_prompt, lin_b, expand_w, gn_w, gn_b,
                                   ec_bf, ep_bf, ep_f, lw_bf, (float*)d_out);
}
